// EncoderBlock_1872605741561
// MI455X (gfx1250) — matrix operand order repaired and checked by reading, not compiled, not run on hardware
//
#include <hip/hip_runtime.h>
#include <hip/hip_bf16.h>
#include <stdint.h>

// ---- problem constants (from reference) ----
#define B_   8
#define T_   1024
#define DM   1024
#define H_   16
#define DK   64
#define DC   256
#define DCR  512
#define DR   32
#define DFF  4096

typedef __attribute__((ext_vector_type(16))) __bf16 v16bf;
typedef __attribute__((ext_vector_type(8)))  __bf16 v8bf;
typedef __attribute__((ext_vector_type(2)))  __bf16 v2bf;
typedef __attribute__((ext_vector_type(8)))  float  v8f;

union Uv16 { v16bf v; uint4 q[2]; };
union Uv8  { v8f v; float f[8]; };
union Upk  { v8bf h; uint4 q; };

// ---- fp32 -> bf16: single vector convert -> v_cvt_pk_bf16_f32 pairs ----
__device__ __forceinline__ uint4 pack8(float4 a, float4 b) {
  v8f f;
  f[0] = a.x; f[1] = a.y; f[2] = a.z; f[3] = a.w;
  f[4] = b.x; f[5] = b.y; f[6] = b.z; f[7] = b.w;
  Upk u;
  u.h = __builtin_convertvector(f, v8bf);
  return u.q;
}
__device__ __forceinline__ uint4 pack8a(const float* f8) {
  v8f f;
#pragma unroll
  for (int i = 0; i < 8; ++i) f[i] = f8[i];
  Upk u;
  u.h = __builtin_convertvector(f, v8bf);
  return u.q;
}
__device__ __forceinline__ unsigned short f2bfs(float f) {
  union { __bf16 h; unsigned short s; } x;
  x.h = (__bf16)f;
  return x.s;
}

// =====================================================================
// Generic GEMM: C[M,N] = act(A[M,K] * W[N,K]^T + bias) (+ residual)
// bf16 WMMA 16x16x32, fp32 accumulate.
// BM=128 BN=64 BK=32, 256 threads (8 waves), wave tile 32x32 (4 WMMAs).
// Software-pipelined: double-buffered LDS, 1 barrier / K-step, last step
// peeled so the steady-state body is branch-free.
// =====================================================================
constexpr int GBM = 128, GBN = 64, GBK = 32;

__global__ __launch_bounds__(256) void gemm_kernel(
    const float* __restrict__ A, const float* __restrict__ W,
    const float* __restrict__ bias, const float* __restrict__ res,
    float* __restrict__ C, int M, int N, int K, int relu)
{
  __shared__ __align__(16) unsigned short As[2][GBM][GBK];
  __shared__ __align__(16) unsigned short Bs[2][GBN][GBK];

  const int tid  = threadIdx.x;
  const int lane = tid & 31, wave = tid >> 5;
  const int m0 = blockIdx.y * GBM, n0 = blockIdx.x * GBN;
  const int wwm = (wave & 3) * 32;     // wave's M offset in tile
  const int wwn = (wave >> 2) * 32;    // wave's N offset in tile
  const int hf = lane >> 4, mr = lane & 15;

  Uv8 c[2][2];
#pragma unroll
  for (int i = 0; i < 2; ++i)
#pragma unroll
    for (int j = 0; j < 2; ++j)
#pragma unroll
      for (int r = 0; r < 8; ++r) c[i][j].f[r] = 0.f;

  const int ar = tid >> 1;             // A staging row 0..127
  const int ac = (tid & 1) * 16;       // A staging col group (16 elems)
  const int br = tid >> 2;             // B staging row 0..63
  const int bc = (tid & 3) * 8;        // B staging col group (8 elems)
  const bool bok = (n0 + br) < N;
  const float* aptr = A + (size_t)(m0 + ar) * K + ac;
  const float* wptr = W + (size_t)(n0 + br) * K + bc;

  float4 ra0, ra1, ra2, ra3, rb0, rb1;

  auto gload = [&](int k0) {
    const float* ap = aptr + k0;
    ra0 = *(const float4*)ap;
    ra1 = *(const float4*)(ap + 4);
    ra2 = *(const float4*)(ap + 8);
    ra3 = *(const float4*)(ap + 12);
    if (bok) {
      const float* wp = wptr + k0;
      rb0 = *(const float4*)wp;
      rb1 = *(const float4*)(wp + 4);
    } else {
      rb0 = make_float4(0.f, 0.f, 0.f, 0.f);
      rb1 = make_float4(0.f, 0.f, 0.f, 0.f);
    }
  };
  auto sstore = [&](int buf) {
    *(uint4*)&As[buf][ar][ac]     = pack8(ra0, ra1);
    *(uint4*)&As[buf][ar][ac + 8] = pack8(ra2, ra3);
    *(uint4*)&Bs[buf][br][bc]     = pack8(rb0, rb1);
  };
  auto compute = [&](int cur) {
    // fragments (ISA 7.12.2 layouts): 2 A-frags, 2 B-frags, 4 WMMAs
    Uv16 af[2], bfr[2];
#pragma unroll
    for (int t = 0; t < 2; ++t) {
      af[t].q[0]  = *(const uint4*)&As[cur][wwm + t * 16 + mr][hf * 8];
      af[t].q[1]  = *(const uint4*)&As[cur][wwm + t * 16 + mr][16 + hf * 8];
      bfr[t].q[0] = *(const uint4*)&Bs[cur][wwn + t * 16 + mr][hf * 8];
      bfr[t].q[1] = *(const uint4*)&Bs[cur][wwn + t * 16 + mr][16 + hf * 8];
    }
#pragma unroll
    for (int ti = 0; ti < 2; ++ti)
#pragma unroll
      for (int tj = 0; tj < 2; ++tj)
        c[ti][tj].v = __builtin_amdgcn_wmma_f32_16x16x32_bf16(
            false, af[ti].v, false, bfr[tj].v, (short)0, c[ti][tj].v, false, false);
  };

  gload(0);
  sstore(0);
  __syncthreads();

  const int nk = K / GBK;
  for (int i = 0; i < nk - 1; ++i) {
    gload((i + 1) * GBK);      // prefetch next tile into regs
    compute(i & 1);            // WMMAs on current buffer
    sstore((i + 1) & 1);       // convert + stage into other buffer
    __syncthreads();
  }
  compute((nk - 1) & 1);       // peeled last step

  // epilogue: C layout -> VGPR r holds rows (r) / (r+8) for lane halves
#pragma unroll
  for (int i = 0; i < 2; ++i) {
    const int rowb = m0 + wwm + i * 16 + (hf ? 8 : 0);
#pragma unroll
    for (int j = 0; j < 2; ++j) {
      const int col = n0 + wwn + j * 16 + mr;
      if (col < N) {
        float bv = bias[col];
#pragma unroll
        for (int r = 0; r < 8; ++r) {
          int row = rowb + r;
          float vv = c[i][j].f[r] + bv;
          if (relu) vv = fmaxf(vv, 0.f);
          if (res) vv += res[(size_t)row * N + col];
          C[(size_t)row * N + col] = vv;
        }
      }
    }
  }
}

// =====================================================================
// LayerNorm over D_MODEL=1024, one block (256 threads) per token.
// =====================================================================
__global__ __launch_bounds__(256) void ln_kernel(
    const float* __restrict__ x, const float* __restrict__ g,
    const float* __restrict__ b, float* __restrict__ y)
{
  __shared__ float rs[8], rq[8];
  const int t = blockIdx.x, tid = threadIdx.x;
  const float* xr = x + (size_t)t * DM;
  float4 v = ((const float4*)xr)[tid];
  float s = v.x + v.y + v.z + v.w;
  float q = v.x * v.x + v.y * v.y + v.z * v.z + v.w * v.w;
#pragma unroll
  for (int o = 16; o >= 1; o >>= 1) {
    s += __shfl_xor(s, o, 32);
    q += __shfl_xor(q, o, 32);
  }
  if ((tid & 31) == 0) { rs[tid >> 5] = s; rq[tid >> 5] = q; }
  __syncthreads();
  float ts = 0.f, tq = 0.f;
#pragma unroll
  for (int i = 0; i < 8; ++i) { ts += rs[i]; tq += rq[i]; }
  const float mu  = ts * (1.f / DM);
  const float var = tq * (1.f / DM) - mu * mu;
  const float inv = rsqrtf(var + 1e-6f);
  float4 gg = ((const float4*)g)[tid];
  float4 bb = ((const float4*)b)[tid];
  float4 o;
  o.x = gg.x * (v.x - mu) * inv + bb.x;
  o.y = gg.y * (v.y - mu) * inv + bb.y;
  o.z = gg.z * (v.z - mu) * inv + bb.z;
  o.w = gg.w * (v.w - mu) * inv + bb.w;
  ((float4*)(y + (size_t)t * DM))[tid] = o;
}

// =====================================================================
// RoPE assembly: overwrite dims [32..63] of each head of q_full/k_full
// with [sin(x[:16]*freq), cos(x[:16]*freq)] of the rope projections.
// =====================================================================
__global__ __launch_bounds__(256) void rope_kernel(
    float* __restrict__ qf, const float* __restrict__ qr,
    float* __restrict__ kf, const float* __restrict__ kr)
{
  int idx = blockIdx.x * 256 + threadIdx.x;   // (token, head, j<16)
  int j = idx & 15;
  int h = (idx >> 4) & (H_ - 1);
  int t = idx >> 8;
  if (t >= B_ * T_) return;
  float freq = __powf(10000.f, -(float)j * (1.f / 16.f));
  float aq = qr[((size_t)t * H_ + h) * DR + j] * freq;
  float ak = kr[(size_t)t * DR + j] * freq;
  float* qd = qf + ((size_t)t * H_ + h) * DK;
  float* kd = kf + ((size_t)t * H_ + h) * DK;
  qd[32 + j] = __sinf(aq);
  qd[48 + j] = __cosf(aq);
  kd[32 + j] = __sinf(ak);
  kd[48 + j] = __cosf(ak);
}

// =====================================================================
// Flash attention: block = (b, h, 128 q rows); 8 waves, wave owns 16 rows.
// Key blocks of 32 tokens; online softmax wave-local; WMMA for QK^T & PV.
// =====================================================================
constexpr int BQ = 128, SK = 32;

__global__ __launch_bounds__(256) void attn_kernel(
    const float* __restrict__ Q, const float* __restrict__ Kg,
    const float* __restrict__ V, const int* __restrict__ mask,
    float* __restrict__ O)
{
  __shared__ __align__(16) unsigned short Kt[SK][DK];      // [token][d]
  __shared__ __align__(16) unsigned short Vt[DK][SK];      // [d][token] (transposed)
  __shared__ __align__(16) unsigned short Ps[8][16][SK];   // per-wave P staging

  const int tid = threadIdx.x, lane = tid & 31, wave = tid >> 5;
  const int hf = lane >> 4, mr = lane & 15;
  const int qt = blockIdx.x, h = blockIdx.y, b = blockIdx.z;
  const int q0 = qt * BQ;
  const int qrow = q0 + wave * 16 + mr;

  // preload Q in A-fragment layout (2 K-steps over d_k = 64)
  const float* qp = Q + ((size_t)(b * T_ + qrow) * H_ + h) * DK;
  Uv16 qa[2];
#pragma unroll
  for (int kc = 0; kc < 2; ++kc) {
    int base = kc * 32 + hf * 16;
    float4 a0 = *(const float4*)(qp + base);
    float4 a1 = *(const float4*)(qp + base + 4);
    float4 a2 = *(const float4*)(qp + base + 8);
    float4 a3 = *(const float4*)(qp + base + 12);
    qa[kc].q[0] = pack8(a0, a1);
    qa[kc].q[1] = pack8(a2, a3);
  }

  Uv8 acc[4];
  float mrun[8], lrun[8];
#pragma unroll
  for (int dt = 0; dt < 4; ++dt)
#pragma unroll
    for (int r = 0; r < 8; ++r) acc[dt].f[r] = 0.f;
#pragma unroll
  for (int r = 0; r < 8; ++r) { mrun[r] = -3.0e38f; lrun[r] = 0.f; }

  const int sr = tid >> 3, sg = (tid & 7) * 8;   // K staging: 32 rows x 8 groups
  const int vd = tid >> 2, vg = (tid & 3) * 8;   // V staging: 64 dims x 4 token groups

  for (int j = 0; j < T_ / SK; ++j) {
    // stage K block
    {
      const float* kp = Kg + ((size_t)(b * T_ + j * SK + sr) * H_ + h) * DK + sg;
      float4 k0 = *(const float4*)kp;
      float4 k1 = *(const float4*)(kp + 4);
      *(uint4*)&Kt[sr][sg] = pack8(k0, k1);
    }
    // stage V block transposed
    {
      float vv[8];
#pragma unroll
      for (int i = 0; i < 8; ++i)
        vv[i] = V[((size_t)(b * T_ + j * SK + vg + i) * H_ + h) * DK + vd];
      *(uint4*)&Vt[vd][vg] = pack8a(vv);
    }
    __syncthreads();

    // S = Q * K^T : two 16x16 tiles (token cols 0..15 and 16..31)
    Uv8 s0, s1;
#pragma unroll
    for (int r = 0; r < 8; ++r) { s0.f[r] = 0.f; s1.f[r] = 0.f; }
#pragma unroll
    for (int kc = 0; kc < 2; ++kc) {
      Uv16 kb0, kb1;
      kb0.q[0] = *(const uint4*)&Kt[mr][kc * 32 + hf * 16];
      kb0.q[1] = *(const uint4*)&Kt[mr][kc * 32 + hf * 16 + 8];
      kb1.q[0] = *(const uint4*)&Kt[16 + mr][kc * 32 + hf * 16];
      kb1.q[1] = *(const uint4*)&Kt[16 + mr][kc * 32 + hf * 16 + 8];
      s0.v = __builtin_amdgcn_wmma_f32_16x16x32_bf16(false, qa[kc].v, false, kb0.v,
                                                     (short)0, s0.v, false, false);
      s1.v = __builtin_amdgcn_wmma_f32_16x16x32_bf16(false, qa[kc].v, false, kb1.v,
                                                     (short)0, s1.v, false, false);
    }

    // scale + mask
    const int tok0 = j * SK + mr, tok1 = tok0 + 16;
    const bool mk0 = mask[b * T_ + tok0] != 0;
    const bool mk1 = mask[b * T_ + tok1] != 0;
#pragma unroll
    for (int r = 0; r < 8; ++r) {
      s0.f[r] = mk0 ? s0.f[r] * 0.125f : -1e9f;
      s1.f[r] = mk1 ? s1.f[r] * 0.125f : -1e9f;
    }

    // online softmax; rows are wave-local, reduce across 16 lanes
    float corr[8];
#pragma unroll
    for (int r = 0; r < 8; ++r) {
      float t = fmaxf(s0.f[r], s1.f[r]);
      t = fmaxf(t, __shfl_xor(t, 1, 32));
      t = fmaxf(t, __shfl_xor(t, 2, 32));
      t = fmaxf(t, __shfl_xor(t, 4, 32));
      t = fmaxf(t, __shfl_xor(t, 8, 32));
      float mnew = fmaxf(mrun[r], t);
      corr[r] = __expf(mrun[r] - mnew);
      mrun[r] = mnew;
      s0.f[r] = __expf(s0.f[r] - mnew);
      s1.f[r] = __expf(s1.f[r] - mnew);
      float rsum = s0.f[r] + s1.f[r];
      rsum += __shfl_xor(rsum, 1, 32);
      rsum += __shfl_xor(rsum, 2, 32);
      rsum += __shfl_xor(rsum, 4, 32);
      rsum += __shfl_xor(rsum, 8, 32);
      lrun[r] = lrun[r] * corr[r] + rsum;
#pragma unroll
      for (int dt = 0; dt < 4; ++dt) acc[dt].f[r] *= corr[r];
    }

    // P (C-layout) -> per-wave LDS -> A-fragment (same-wave DS is in-order)
#pragma unroll
    for (int r = 0; r < 8; ++r) {
      int m = r + hf * 8;
      Ps[wave][m][mr]      = f2bfs(s0.f[r]);
      Ps[wave][m][mr + 16] = f2bfs(s1.f[r]);
    }
    Uv16 pa;
    pa.q[0] = *(const uint4*)&Ps[wave][mr][hf * 16];
    pa.q[1] = *(const uint4*)&Ps[wave][mr][hf * 16 + 8];

#pragma unroll
    for (int dt = 0; dt < 4; ++dt) {
      Uv16 vb;
      vb.q[0] = *(const uint4*)&Vt[dt * 16 + mr][hf * 16];
      vb.q[1] = *(const uint4*)&Vt[dt * 16 + mr][hf * 16 + 8];
      acc[dt].v = __builtin_amdgcn_wmma_f32_16x16x32_bf16(false, pa.v, false, vb.v,
                                                          (short)0, acc[dt].v, false, false);
    }
    __syncthreads();
  }

  // normalize and store [B,T,H,DK] == [B,T,DM]
  float rl[8];
#pragma unroll
  for (int r = 0; r < 8; ++r) rl[r] = 1.0f / lrun[r];
  const int orow = q0 + wave * 16 + hf * 8;
#pragma unroll
  for (int dt = 0; dt < 4; ++dt) {
    int d = dt * 16 + mr;
#pragma unroll
    for (int r = 0; r < 8; ++r)
      O[((size_t)(b * T_ + orow + r) * H_ + h) * DK + d] = acc[dt].f[r] * rl[r];
  }
}

// =====================================================================
extern "C" void kernel_launch(void* const* d_in, const int* in_sizes, int n_in,
                              void* d_out, int out_size, void* d_ws, size_t ws_size,
                              hipStream_t stream) {
  (void)in_sizes; (void)n_in; (void)out_size; (void)ws_size;
  const float* x      = (const float*)d_in[0];
  const int*   mask   = (const int*)d_in[1];
  const float* W_DKV_w = (const float*)d_in[2];
  const float* W_DKV_b = (const float*)d_in[3];
  const float* W_UK_w  = (const float*)d_in[4];
  const float* W_UK_b  = (const float*)d_in[5];
  const float* W_UV_w  = (const float*)d_in[6];
  const float* W_UV_b  = (const float*)d_in[7];
  const float* W_DQ_w  = (const float*)d_in[8];
  const float* W_DQ_b  = (const float*)d_in[9];
  const float* W_UQ_w  = (const float*)d_in[10];
  const float* W_UQ_b  = (const float*)d_in[11];
  const float* W_QR_w  = (const float*)d_in[12];
  const float* W_QR_b  = (const float*)d_in[13];
  const float* W_KR_w  = (const float*)d_in[14];
  const float* W_KR_b  = (const float*)d_in[15];
  const float* W_O_w   = (const float*)d_in[16];
  const float* W_O_b   = (const float*)d_in[17];
  const float* ff1_w   = (const float*)d_in[18];
  const float* ff1_b   = (const float*)d_in[19];
  const float* ff2_w   = (const float*)d_in[20];
  const float* ff2_b   = (const float*)d_in[21];
  const float* ln1_g   = (const float*)d_in[22];
  const float* ln1_b   = (const float*)d_in[23];
  const float* ln2_g   = (const float*)d_in[24];
  const float* ln2_b   = (const float*)d_in[25];

  const size_t BT = (size_t)B_ * T_;
  float* ws   = (float*)d_ws;
  float* xln  = ws;                      // BT*DM
  float* ckv  = xln  + BT * DM;          // BT*DC
  float* kful = ckv  + BT * DC;          // BT*DM
  float* vbuf = kful + BT * DM;          // BT*DM
  float* cq   = vbuf + BT * DM;          // BT*DCR
  float* qful = cq   + BT * DCR;         // BT*DM
  float* qrp  = qful + BT * DM;          // BT*H*DR
  float* krp  = qrp  + BT * H_ * DR;     // BT*DR
  float* attn = krp  + BT * DR;          // BT*DM
  float* hbuf = attn + BT * DM;          // BT*DM
  float* hln  = xln;                     // reuse (dead after attention)
  float* ffh  = ckv;                     // reuse 128MB region (dead by FFN)

  dim3 blk(256);
  auto gemm = [&](const float* A, const float* W, const float* bias,
                  const float* res, float* C, int M, int N, int K, int relu) {
    dim3 g((N + GBN - 1) / GBN, M / GBM);
    gemm_kernel<<<g, blk, 0, stream>>>(A, W, bias, res, C, M, N, K, relu);
  };

  ln_kernel<<<(int)BT, blk, 0, stream>>>(x, ln1_g, ln1_b, xln);

  gemm(xln, W_DKV_w, W_DKV_b, nullptr, ckv,  (int)BT, DC,       DM,  0);
  gemm(ckv, W_UK_w,  W_UK_b,  nullptr, kful, (int)BT, DM,       DC,  0);
  gemm(ckv, W_UV_w,  W_UV_b,  nullptr, vbuf, (int)BT, DM,       DC,  0);
  gemm(xln, W_DQ_w,  W_DQ_b,  nullptr, cq,   (int)BT, DCR,      DM,  0);
  gemm(cq,  W_UQ_w,  W_UQ_b,  nullptr, qful, (int)BT, DM,       DCR, 0);
  gemm(cq,  W_QR_w,  W_QR_b,  nullptr, qrp,  (int)BT, H_ * DR,  DCR, 0);
  gemm(xln, W_KR_w,  W_KR_b,  nullptr, krp,  (int)BT, DR,       DM,  0);

  rope_kernel<<<(int)((BT * H_ * 16 + 255) / 256), blk, 0, stream>>>(qful, qrp, kful, krp);

  attn_kernel<<<dim3(T_ / BQ, H_, B_), blk, 0, stream>>>(qful, kful, vbuf, mask, attn);

  gemm(attn, W_O_w, W_O_b, x, hbuf, (int)BT, DM, DM, 0);

  ln_kernel<<<(int)BT, blk, 0, stream>>>(hbuf, ln2_g, ln2_b, hln);

  gemm(hln, ff1_w, ff1_b, nullptr, ffh,          (int)BT, DFF, DM,  1);
  gemm(ffh, ff2_w, ff2_b, hbuf,    (float*)d_out,(int)BT, DM,  DFF, 0);
}
